// Phase2_TransformerDecoder_24489903522110
// MI455X (gfx1250) — compile-verified, matrix operand order checked by reading, not run on hardware
//
#include <hip/hip_runtime.h>
#include <cstddef>

// Problem constants (match the reference)
#define D_MODEL 512
#define NHEAD   8
#define DKH     64
#define BATCH   8
#define TQ      512
#define S_SRC   2048
#define E_ENT   256
#define P_P1    512
#define F_FF    2048

typedef __attribute__((ext_vector_type(16))) __bf16 v16bf;
typedef __attribute__((ext_vector_type(8)))  __bf16 v8bf;
typedef __attribute__((ext_vector_type(4)))  __bf16 v4bf;
typedef __attribute__((ext_vector_type(8)))  float  v8f;
typedef __attribute__((ext_vector_type(4)))  float  v4f;

union BF16Frag { v16bf v; v8bf h[2]; };

// chunk swizzle: distributes 16B chunks across LDS banks for 64B-strided rows
__device__ __forceinline__ int swz4(int r) { return (r ^ (r >> 2)) & 3; }

// branchless tanh-form GELU (no exec-mask dance from libm tanhf)
__device__ __forceinline__ float gelu_tanh(float x) {
    float u = 0.7978845608028654f * (x + 0.044715f * x * x * x);
    float t = 1.0f - 2.0f / (1.0f + __expf(2.0f * u));   // tanh(u)
    return 0.5f * x * (1.0f + t);
}

// ---------------------------------------------------------------------------
// LayerNorm: one block (256 threads) per row of D_MODEL=512
// ---------------------------------------------------------------------------
__global__ __launch_bounds__(256)
void ln_kernel(const float* __restrict__ x, const float* __restrict__ g,
               const float* __restrict__ b, float* __restrict__ y) {
    __shared__ float red[256];
    const int row = blockIdx.x;
    const int t = threadIdx.x;
    const float* xr = x + (size_t)row * D_MODEL;
    float v0 = xr[t], v1 = xr[t + 256];

    red[t] = v0 + v1;
    __syncthreads();
    for (int s = 128; s > 0; s >>= 1) {
        if (t < s) red[t] += red[t + s];
        __syncthreads();
    }
    float mean = red[0] * (1.0f / D_MODEL);
    __syncthreads();

    float d0 = v0 - mean, d1 = v1 - mean;
    red[t] = d0 * d0 + d1 * d1;
    __syncthreads();
    for (int s = 128; s > 0; s >>= 1) {
        if (t < s) red[t] += red[t + s];
        __syncthreads();
    }
    float inv = rsqrtf(red[0] * (1.0f / D_MODEL) + 1e-6f);

    float* yr = y + (size_t)row * D_MODEL;
    yr[t]       = d0 * inv * g[t]       + b[t];
    yr[t + 256] = d1 * inv * g[t + 256] + b[t + 256];
}

// ---------------------------------------------------------------------------
// Gated blend: out = sig(a.w[:D] + b.w[D:] + bias) * a + (1-g) * b (+ resid)
// ---------------------------------------------------------------------------
__global__ __launch_bounds__(256)
void gate_kernel(const float* __restrict__ a, const float* __restrict__ bb,
                 const float* __restrict__ w, const float* __restrict__ bias,
                 const float* __restrict__ resid, float* __restrict__ out) {
    __shared__ float red[256];
    const int row = blockIdx.x;
    const int t = threadIdx.x;
    const float* ar = a + (size_t)row * D_MODEL;
    const float* br = bb + (size_t)row * D_MODEL;
    float a0 = ar[t], a1 = ar[t + 256];
    float b0 = br[t], b1 = br[t + 256];

    red[t] = a0 * w[t] + a1 * w[t + 256] + b0 * w[D_MODEL + t] + b1 * w[D_MODEL + t + 256];
    __syncthreads();
    for (int s = 128; s > 0; s >>= 1) {
        if (t < s) red[t] += red[t + s];
        __syncthreads();
    }
    float gate = 1.0f / (1.0f + __expf(-(red[0] + bias[0])));

    float r0 = resid ? resid[(size_t)row * D_MODEL + t] : 0.0f;
    float r1 = resid ? resid[(size_t)row * D_MODEL + t + 256] : 0.0f;
    out[(size_t)row * D_MODEL + t]       = gate * a0 + (1.0f - gate) * b0 + r0;
    out[(size_t)row * D_MODEL + t + 256] = gate * a1 + (1.0f - gate) * b1 + r1;
}

// ---------------------------------------------------------------------------
// Pack A[M][K] (f32) into bf16 tiles [mb][kb][128*32] whose byte image equals
// the swizzled LDS tile used by the GEMM. One block per tile.
// ---------------------------------------------------------------------------
__global__ __launch_bounds__(256)
void pack_a_kernel(const float* __restrict__ A, __bf16* __restrict__ Ap,
                   int M, int K) {
    const int kb = blockIdx.x, mb = blockIdx.y;
    const int nkb = gridDim.x;
    __bf16* dst = Ap + ((size_t)mb * nkb + kb) * 4096;
    const int tid = threadIdx.x;
#pragma unroll
    for (int i = 0; i < 4; i++) {
        int e = (tid + 256 * i) * 4;
        int r = e >> 5, c = e & 31;   // c multiple of 4
        v4f a = *(const v4f*)&A[(size_t)(mb * 128 + r) * K + kb * 32 + c];
        v4bf ab;
        ab[0] = (__bf16)a[0]; ab[1] = (__bf16)a[1];
        ab[2] = (__bf16)a[2]; ab[3] = (__bf16)a[3];
        *(v4bf*)&dst[r * 32 + (((c >> 3) ^ swz4(r)) * 8) + (c & 7)] = ab;
    }
}

// ---------------------------------------------------------------------------
// Pack W[K][N] (f32) into transposed bf16 tiles [nb][kb][128 cols * 32 k]
// (swizzled LDS image). Loads are coalesced along N; transpose done here once.
// ---------------------------------------------------------------------------
__global__ __launch_bounds__(256)
void pack_b_kernel(const float* __restrict__ W, __bf16* __restrict__ Bp,
                   int K, int N) {
    const int kb = blockIdx.x, nb = blockIdx.y;
    const int nkb = gridDim.x;
    __bf16* dst = Bp + ((size_t)nb * nkb + kb) * 4096;
    const int tid = threadIdx.x;
    const int cc  = tid & 127;
    const int kg0 = tid >> 7;      // 0..1
#pragma unroll
    for (int i = 0; i < 4; i++) {
        int k4 = (kg0 + 2 * i) * 4;   // 0..28, 4 consecutive k in one chunk
        v4bf pb;
#pragma unroll
        for (int j = 0; j < 4; j++)
            pb[j] = (__bf16)W[(size_t)(kb * 32 + k4 + j) * N + nb * 128 + cc];
        *(v4bf*)&dst[cc * 32 + (((k4 >> 3) ^ swz4(cc)) * 8) + (k4 & 7)] = pb;
    }
}

// issue one 8KB tile copy as 2 async b128 loads per thread (4 per wave-pair)
__device__ __forceinline__ void async_copy_8k(const __bf16* gsrc, unsigned ldsDst,
                                              int tid) {
    unsigned long long g = (unsigned long long)(size_t)gsrc + (unsigned)tid * 16u;
    unsigned l = ldsDst + (unsigned)tid * 16u;
    asm volatile("global_load_async_to_lds_b128 %0, %1, off"
                 :: "v"(l), "v"(g) : "memory");
    g += 4096; l += 4096;
    asm volatile("global_load_async_to_lds_b128 %0, %1, off"
                 :: "v"(l), "v"(g) : "memory");
}

// ---------------------------------------------------------------------------
// GEMM on packed bf16 tiles: C[M,N] = act(A@W + bias (+ resid)).
// Block tile 128x128, 8 waves (4Mx2N). Double-buffered LDS, tiles staged with
// GLOBAL_LOAD_ASYNC_TO_LDS_B128 (ASYNCcnt) — prefetch k+1 while computing k.
// All 12 fragment ds_load_b128 issued before the 8 WMMAs for partial waits.
// ---------------------------------------------------------------------------
__global__ __launch_bounds__(256)
void gemm_pk_kernel(const __bf16* __restrict__ Ap, const __bf16* __restrict__ Bp,
                    const float* __restrict__ bias, const float* __restrict__ resid,
                    float* __restrict__ C, int M, int N, int K, int act) {
    __shared__ alignas(16) __bf16 sm[2][8192];   // [buf][A tile | B tile]
    const int tid  = threadIdx.x;
    const int lane = tid & 31;
    const int wave = tid >> 5;
    const int wm = wave >> 1;   // 0..3
    const int wn = wave & 1;    // 0..1
    const int ln = lane & 15;
    const int kh = lane >> 4;
    const int m0 = blockIdx.y * 128;
    const int n0 = blockIdx.x * 128;
    const int nkb = K >> 5;

    // low 32 bits of a generic LDS pointer == LDS byte offset (aperture scheme)
    const unsigned smBase = (unsigned)(size_t)(void*)&sm[0][0];

    const __bf16* Atiles = Ap + (size_t)blockIdx.y * nkb * 4096;
    const __bf16* Btiles = Bp + (size_t)blockIdx.x * nkb * 4096;

    v8f acc[2][4];
    for (int i = 0; i < 2; i++)
        for (int j = 0; j < 4; j++)
            for (int e = 0; e < 8; e++) acc[i][j][e] = 0.0f;

    // prologue: stage tile 0 into buffer 0
    async_copy_8k(Atiles, smBase, tid);
    async_copy_8k(Btiles, smBase + 4096 * 2, tid);

    for (int kb = 0; kb < nkb; kb++) {
        const int cur = kb & 1;
        const bool hasNext = (kb + 1) < nkb;
        if (hasNext) {
            unsigned nb = smBase + (cur ^ 1) * 16384u;
            async_copy_8k(Atiles + (size_t)(kb + 1) * 4096, nb, tid);
            async_copy_8k(Btiles + (size_t)(kb + 1) * 4096, nb + 8192u, tid);
            asm volatile("s_wait_asynccnt 0x4" ::: "memory");  // current tile done
        } else {
            asm volatile("s_wait_asynccnt 0x0" ::: "memory");
        }
        __syncthreads();   // all waves' async portions visible

        const __bf16* Abase = &sm[cur][0];
        const __bf16* Bbase = &sm[cur][4096];

        // ---- load ALL fragments first (12x ds_load_b128), then 8 WMMAs ----
        BF16Frag af[2], bfr[4];
#pragma unroll
        for (int mf = 0; mf < 2; mf++) {
            int r = wm * 32 + mf * 16 + ln;
            const __bf16* ap = Abase + r * 32;
            int sw = swz4(r);
            af[mf].h[0] = *(const v8bf*)&ap[((kh)     ^ sw) * 8];
            af[mf].h[1] = *(const v8bf*)&ap[((2 + kh) ^ sw) * 8];
        }
#pragma unroll
        for (int nf = 0; nf < 4; nf++) {
            int cN = wn * 64 + nf * 16 + ln;
            const __bf16* bp = Bbase + cN * 32;
            int sw = swz4(cN);
            bfr[nf].h[0] = *(const v8bf*)&bp[((kh)     ^ sw) * 8];
            bfr[nf].h[1] = *(const v8bf*)&bp[((2 + kh) ^ sw) * 8];
        }
#pragma unroll
        for (int nf = 0; nf < 4; nf++) {
            acc[0][nf] = __builtin_amdgcn_wmma_f32_16x16x32_bf16(
                false, af[0].v, false, bfr[nf].v, (short)0, acc[0][nf], false, false);
            acc[1][nf] = __builtin_amdgcn_wmma_f32_16x16x32_bf16(
                false, af[1].v, false, bfr[nf].v, (short)0, acc[1][nf], false, false);
        }
        __syncthreads();   // reads done before next iteration's async overwrite
    }

    // ---- epilogue (branches hoisted; uniform per kernel launch) ----
    if (act) {
#pragma unroll
        for (int mf = 0; mf < 2; mf++)
#pragma unroll
            for (int nf = 0; nf < 4; nf++) {
                int col = n0 + wn * 64 + nf * 16 + ln;
                float bcol = bias[col];
#pragma unroll
                for (int r = 0; r < 8; r++) {
                    int row = m0 + wm * 32 + mf * 16 + r + kh * 8;
                    C[(size_t)row * N + col] = gelu_tanh(acc[mf][nf][r] + bcol);
                }
            }
    } else if (resid) {
#pragma unroll
        for (int mf = 0; mf < 2; mf++)
#pragma unroll
            for (int nf = 0; nf < 4; nf++) {
                int col = n0 + wn * 64 + nf * 16 + ln;
                float bcol = bias[col];
#pragma unroll
                for (int r = 0; r < 8; r++) {
                    int row = m0 + wm * 32 + mf * 16 + r + kh * 8;
                    C[(size_t)row * N + col] =
                        acc[mf][nf][r] + bcol + resid[(size_t)row * N + col];
                }
            }
    } else {
#pragma unroll
        for (int mf = 0; mf < 2; mf++)
#pragma unroll
            for (int nf = 0; nf < 4; nf++) {
                int col = n0 + wn * 64 + nf * 16 + ln;
                float bcol = bias[col];
#pragma unroll
                for (int r = 0; r < 8; r++) {
                    int row = m0 + wm * 32 + mf * 16 + r + kh * 8;
                    C[(size_t)row * N + col] = acc[mf][nf][r] + bcol;
                }
            }
    }
}

// helper: build a bf16 A/B fragment from 4 v4f (b128) loads, optional scale
__device__ __forceinline__ v16bf frag_from_f32(const float* p, int kh, float scale) {
    v4f q0 = *(const v4f*)(p + kh * 8);
    v4f q1 = *(const v4f*)(p + kh * 8 + 4);
    v4f q2 = *(const v4f*)(p + 16 + kh * 8);
    v4f q3 = *(const v4f*)(p + 16 + kh * 8 + 4);
    v16bf f;
#pragma unroll
    for (int j = 0; j < 4; j++) {
        f[j]      = (__bf16)(q0[j] * scale);
        f[4 + j]  = (__bf16)(q1[j] * scale);
        f[8 + j]  = (__bf16)(q2[j] * scale);
        f[12 + j] = (__bf16)(q3[j] * scale);
    }
    return f;
}

// ---------------------------------------------------------------------------
// Fused attention (flash-style). Grid: (TQ/64, NHEAD, BATCH), block = 128.
// ---------------------------------------------------------------------------
__global__ __launch_bounds__(128)
void attn_kernel(const float* __restrict__ Q, const float* __restrict__ K,
                 const float* __restrict__ V, float* __restrict__ O,
                 const unsigned char* __restrict__ mask,
                 const float* __restrict__ score, int Skv, int causal) {
    __shared__ alignas(16) float sP[4][16][32];
    __shared__ float sM[4][16], sL[4][16], sC[4][16];
    const int lane = threadIdx.x & 31;
    const int w    = threadIdx.x >> 5;
    const int ln = lane & 15;
    const int kh = lane >> 4;
    const int b = blockIdx.z, h = blockIdx.y;
    const int qbase = (blockIdx.x * 4 + w) * 16;

    if (lane < 16) { sM[w][lane] = -1e30f; sL[w][lane] = 0.0f; }

    v8f acc[4];
    for (int g = 0; g < 4; g++)
        for (int e = 0; e < 8; e++) acc[g][e] = 0.0f;

    const float* qr = Q + (size_t)(b * TQ + qbase + ln) * D_MODEL + h * DKH;
    v16bf aq0 = frag_from_f32(qr,      kh, 0.125f);
    v16bf aq1 = frag_from_f32(qr + 32, kh, 0.125f);

    const int nkt = Skv >> 5;
    for (int kt = 0; kt < nkt; kt++) {
#pragma unroll
        for (int t = 0; t < 2; t++) {
            int key = kt * 32 + t * 16 + ln;
            const float* kr = K + (size_t)(b * Skv + key) * D_MODEL + h * DKH;
            v16bf bk0 = frag_from_f32(kr,      kh, 1.0f);
            v16bf bk1 = frag_from_f32(kr + 32, kh, 1.0f);
            v8f s;
            for (int e = 0; e < 8; e++) s[e] = 0.0f;
            s = __builtin_amdgcn_wmma_f32_16x16x32_bf16(
                false, aq0, false, bk0, (short)0, s, false, false);
            s = __builtin_amdgcn_wmma_f32_16x16x32_bf16(
                false, aq1, false, bk1, (short)0, s, false, false);
#pragma unroll
            for (int r = 0; r < 8; r++) {
                int ml = r + kh * 8;
                int qg = qbase + ml;
                float val = s[r];
                if (causal && key > qg) val = -1e30f;
                if (mask && mask[((size_t)b * TQ + qg) * Skv + key]) val = -1e30f;
                sP[w][ml][t * 16 + ln] = val;
            }
        }
        __syncthreads();

        if (lane < 16) {
            float mold = sM[w][lane];
            float mt = mold;
#pragma unroll 4
            for (int j = 0; j < 32; j++) mt = fmaxf(mt, sP[w][lane][j]);
            float corr = __expf(mold - mt);
            float sum = 0.0f;
            for (int j = 0; j < 32; j++) {
                float p = __expf(sP[w][lane][j] - mt);
                if (score) p *= score[(size_t)b * Skv + kt * 32 + j];
                sP[w][lane][j] = p;
                sum += p;
            }
            sM[w][lane] = mt;
            sL[w][lane] = sL[w][lane] * corr + sum;
            sC[w][lane] = corr;
        }
        __syncthreads();

        v16bf ap = frag_from_f32(&sP[w][ln][0], kh, 1.0f);
#pragma unroll
        for (int g = 0; g < 4; g++) {
#pragma unroll
            for (int r = 0; r < 8; r++) acc[g][r] *= sC[w][r + kh * 8];
            int col = h * DKH + g * 16 + ln;
            v16bf bv;
#pragma unroll
            for (int j = 0; j < 8; j++) {
                bv[j]     = (__bf16)V[(size_t)(b * Skv + kt * 32 + kh * 8 + j) * D_MODEL + col];
                bv[8 + j] = (__bf16)V[(size_t)(b * Skv + kt * 32 + 16 + kh * 8 + j) * D_MODEL + col];
            }
            acc[g] = __builtin_amdgcn_wmma_f32_16x16x32_bf16(
                false, ap, false, bv, (short)0, acc[g], false, false);
        }
        __syncthreads();
    }

    if (lane < 16) sC[w][lane] = 1.0f / (sL[w][lane] + (score ? 1e-9f : 0.0f));
    __syncthreads();
#pragma unroll
    for (int g = 0; g < 4; g++)
#pragma unroll
        for (int r = 0; r < 8; r++) {
            int ml = r + kh * 8;
            O[(size_t)(b * TQ + qbase + ml) * D_MODEL + h * DKH + g * 16 + ln] =
                acc[g][r] * sC[w][ml];
        }
}

// ---------------------------------------------------------------------------
// Host-side orchestration
// ---------------------------------------------------------------------------
static __bf16* g_Ap = nullptr;
static __bf16* g_Bp = nullptr;

static inline void launch_gemm(const float* A, const float* W, const float* bias,
                               const float* resid, float* C, int M, int N, int K,
                               int act, hipStream_t s) {
    pack_a_kernel<<<dim3(K / 32, M / 128), 256, 0, s>>>(A, g_Ap, M, K);
    pack_b_kernel<<<dim3(K / 32, N / 128), 256, 0, s>>>(W, g_Bp, K, N);
    gemm_pk_kernel<<<dim3(N / 128, M / 128), 256, 0, s>>>(g_Ap, g_Bp, bias, resid,
                                                          C, M, N, K, act);
}

static inline void launch_attn(const float* Q, const float* K, const float* V,
                               float* O, const unsigned char* mask,
                               const float* score, int Skv, int causal,
                               hipStream_t s) {
    dim3 grid(TQ / 64, NHEAD, BATCH);
    attn_kernel<<<grid, 128, 0, s>>>(Q, K, V, O, mask, score, Skv, causal);
}

extern "C" void kernel_launch(void* const* d_in, const int* in_sizes, int n_in,
                              void* d_out, int out_size, void* d_ws, size_t ws_size,
                              hipStream_t stream) {
    (void)in_sizes; (void)n_in; (void)out_size; (void)ws_size;
    const int D = D_MODEL;

    const float* inputs        = (const float*)d_in[0];
    const float* memory_bank   = (const float*)d_in[1];
    const float* ent_context   = (const float*)d_in[2];
    const float* ent_score     = (const float*)d_in[3];
    const float* phase1_memory = (const float*)d_in[4];
    const unsigned char* src_pad = (const unsigned char*)d_in[5];
    const unsigned char* tgt_pad = (const unsigned char*)d_in[6];
    const unsigned char* ent_pad = (const unsigned char*)d_in[7];
    const unsigned char* p1_pad  = (const unsigned char*)d_in[8];

    const float* self_Wqkv  = (const float*)d_in[9];
    const float* self_bqkv  = (const float*)d_in[10];
    const float* self_Wo    = (const float*)d_in[11];
    const float* self_bo    = (const float*)d_in[12];
    const float* word_Wqkv  = (const float*)d_in[13];
    const float* word_bqkv  = (const float*)d_in[14];
    const float* word_Wo    = (const float*)d_in[15];
    const float* word_bo    = (const float*)d_in[16];
    const float* graph_Wqkv = (const float*)d_in[17];
    const float* graph_bqkv = (const float*)d_in[18];
    const float* graph_Wo   = (const float*)d_in[19];
    const float* graph_bo   = (const float*)d_in[20];
    const float* p1_Wqkv    = (const float*)d_in[21];
    const float* p1_bqkv    = (const float*)d_in[22];
    const float* p1_Wo      = (const float*)d_in[23];
    const float* p1_bo      = (const float*)d_in[24];

    const float* ln1_g = (const float*)d_in[25];
    const float* ln1_b = (const float*)d_in[26];
    const float* ln2_g = (const float*)d_in[27];
    const float* ln2_b = (const float*)d_in[28];
    const float* ffn_ln_g = (const float*)d_in[29];
    const float* ffn_ln_b = (const float*)d_in[30];
    const float* fus1_w = (const float*)d_in[31];
    const float* fus1_b = (const float*)d_in[32];
    const float* fus2_w = (const float*)d_in[33];
    const float* fus2_b = (const float*)d_in[34];
    const float* ffn_w1 = (const float*)d_in[35];
    const float* ffn_b1 = (const float*)d_in[36];
    const float* ffn_w2 = (const float*)d_in[37];
    const float* ffn_b2 = (const float*)d_in[38];

    const size_t NT = (size_t)BATCH * TQ;    // 4096 query tokens
    const size_t NS = (size_t)BATCH * S_SRC; // 16384 memory tokens (largest K/V)
    const size_t NE = (size_t)BATCH * E_ENT; // 2048
    const size_t NP = (size_t)BATCH * P_P1;  // 4096

    float* ws = (float*)d_ws;
    float* xln    = ws;                         // NT*D  (also qn / ffn ln)
    float* bufQ   = xln    + NT * D;            // NT*D
    float* bufCtx = bufQ   + NT * D;            // NT*D
    float* wordO  = bufCtx + NT * D;            // NT*D
    float* gbO    = wordO  + NT * D;            // NT*D  (graph, then p1)
    float* query  = gbO    + NT * D;            // NT*D
    float* outb   = query  + NT * D;            // NT*D
    float* ctxbl  = outb   + NT * D;            // NT*D
    float* bufK   = ctxbl  + NT * D;            // NS*D
    float* bufV   = bufK   + NS * D;            // NS*D
    float* bufH   = bufV   + NS * D;            // NT*F_FF
    g_Ap = (__bf16*)(bufH + NT * F_FF);         // max M*K bf16 = 8.39M elems
    g_Bp = g_Ap + (size_t)16384 * 512;          // max K*N bf16 = 1.05M elems

    // ---- self attention (pre-norm) ----
    ln_kernel<<<(int)NT, 256, 0, stream>>>(inputs, ln1_g, ln1_b, xln);
    launch_gemm(xln, self_Wqkv,             self_bqkv,         nullptr, bufQ, NT, D, D, 0, stream);
    launch_gemm(xln, self_Wqkv + D * D,     self_bqkv + D,     nullptr, bufK, NT, D, D, 0, stream);
    launch_gemm(xln, self_Wqkv + 2 * D * D, self_bqkv + 2 * D, nullptr, bufV, NT, D, D, 0, stream);
    launch_attn(bufQ, bufK, bufV, bufCtx, tgt_pad, nullptr, TQ, 1, stream);
    launch_gemm(bufCtx, self_Wo, self_bo, inputs, query, NT, D, D, 0, stream); // + residual

    ln_kernel<<<(int)NT, 256, 0, stream>>>(query, ln2_g, ln2_b, xln); // qn

    // ---- word context ----
    launch_gemm(xln,         word_Wqkv,             word_bqkv,         nullptr, bufQ, NT, D, D, 0, stream);
    launch_gemm(memory_bank, word_Wqkv + D * D,     word_bqkv + D,     nullptr, bufK, NS, D, D, 0, stream);
    launch_gemm(memory_bank, word_Wqkv + 2 * D * D, word_bqkv + 2 * D, nullptr, bufV, NS, D, D, 0, stream);
    launch_attn(bufQ, bufK, bufV, bufCtx, src_pad, nullptr, S_SRC, 0, stream);
    launch_gemm(bufCtx, word_Wo, word_bo, nullptr, wordO, NT, D, D, 0, stream);

    // ---- graph context (score-reweighted) ----
    launch_gemm(xln,         graph_Wqkv,             graph_bqkv,         nullptr, bufQ, NT, D, D, 0, stream);
    launch_gemm(ent_context, graph_Wqkv + D * D,     graph_bqkv + D,     nullptr, bufK, NE, D, D, 0, stream);
    launch_gemm(ent_context, graph_Wqkv + 2 * D * D, graph_bqkv + 2 * D, nullptr, bufV, NE, D, D, 0, stream);
    launch_attn(bufQ, bufK, bufV, bufCtx, ent_pad, ent_score, E_ENT, 0, stream);
    launch_gemm(bufCtx, graph_Wo, graph_bo, nullptr, gbO, NT, D, D, 0, stream);

    // ---- gate 1: ctx = g1*word + (1-g1)*graph ----
    gate_kernel<<<(int)NT, 256, 0, stream>>>(wordO, gbO, fus1_w, fus1_b, nullptr, ctxbl);

    // ---- phase1 context ----
    launch_gemm(xln,           p1_Wqkv,             p1_bqkv,         nullptr, bufQ, NT, D, D, 0, stream);
    launch_gemm(phase1_memory, p1_Wqkv + D * D,     p1_bqkv + D,     nullptr, bufK, NP, D, D, 0, stream);
    launch_gemm(phase1_memory, p1_Wqkv + 2 * D * D, p1_bqkv + 2 * D, nullptr, bufV, NP, D, D, 0, stream);
    launch_attn(bufQ, bufK, bufV, bufCtx, p1_pad, nullptr, P_P1, 0, stream);
    launch_gemm(bufCtx, p1_Wo, p1_bo, nullptr, gbO, NT, D, D, 0, stream);

    // ---- gate 2 + query residual: out = g2*ctx + (1-g2)*p1 + query ----
    gate_kernel<<<(int)NT, 256, 0, stream>>>(ctxbl, gbO, fus2_w, fus2_b, query, outb);

    // ---- FFN (pre-norm, gelu, residual) ----
    ln_kernel<<<(int)NT, 256, 0, stream>>>(outb, ffn_ln_g, ffn_ln_b, xln);
    launch_gemm(xln,  ffn_w1, ffn_b1, nullptr, bufH,           NT, F_FF, D,    1, stream);
    launch_gemm(bufH, ffn_w2, ffn_b2, outb,    (float*)d_out,  NT, D,    F_FF, 0, stream);
}
